// MultiHeadAttention_77610059039245
// MI455X (gfx1250) — matrix operand order repaired and checked by reading, not compiled, not run on hardware
//
#include <hip/hip_runtime.h>
#include <hip/hip_bf16.h>
#include <stdint.h>

// ---------------------------------------------------------------------------
// MultiHeadAttention forward for MI455X (gfx1250, wave32, WMMA bf16)
//   1) convert X / W to bf16 once (native v_cvt path)
//   2) q,k,v = X*W  (WMMA GEMM, 128x128 block tile, async-LDS staging)
//   3) causal flash attention (WMMA QK^T + PV, online softmax)
//   4) out = attn * Wo (fp32 result)
// ---------------------------------------------------------------------------

typedef __attribute__((ext_vector_type(16))) __bf16 bf16x16;
typedef __attribute__((ext_vector_type(8)))  __bf16 bf16x8;
typedef __attribute__((ext_vector_type(4)))  __bf16 bf16x4;
typedef __attribute__((ext_vector_type(8)))  float  v8f;
typedef int i32x4 __attribute__((vector_size(4 * sizeof(int))));

#define DEVFN static __device__ __forceinline__

constexpr int Bn = 4;
constexpr int Sn = 2048;
constexpr int En = 1024;
constexpr int Hn = 16;
constexpr int Dh = 64;

// ---- gfx1250 async global->LDS copy (ASYNCcnt path), with sync fallback ----
#if defined(__has_builtin)
#if __has_builtin(__builtin_amdgcn_global_load_async_to_lds_b128)
#define HAVE_ASYNC_LDS 1
#endif
#endif

DEVFN void copy16_g2l(void* lds_dst, const void* gsrc) {
#if defined(HAVE_ASYNC_LDS)
  __builtin_amdgcn_global_load_async_to_lds_b128(
      (__attribute__((address_space(1))) i32x4*)(void*)gsrc,
      (__attribute__((address_space(3))) i32x4*)lds_dst, 0, 0);
#else
  *(uint4*)lds_dst = *(const uint4*)gsrc;
#endif
}
DEVFN void async_wait0() {
#if defined(HAVE_ASYNC_LDS)
#if __has_builtin(__builtin_amdgcn_s_wait_asynccnt)
  __builtin_amdgcn_s_wait_asynccnt(0);
#else
  asm volatile("s_wait_asynccnt 0x0" ::: "memory");
#endif
#endif
}

DEVFN bf16x16 frag_cat(bf16x8 lo, bf16x8 hi) {
  return __builtin_shufflevector(lo, hi, 0,1,2,3,4,5,6,7,8,9,10,11,12,13,14,15);
}

// A-fragment, 16x32 bf16 (ISA 7.12.2): lane holds row M=lane&15.
// lanes 0-15: K 0..7 then 16..23 ; lanes 16-31: K 8..15 then 24..31.
DEVFN bf16x16 load_frag_a(const __bf16* tile, int ldm, int lane) {
  int row  = lane & 15;
  int koff = (lane & 16) ? 8 : 0;
  const __bf16* p = tile + row * ldm + koff;
  return frag_cat(*(const bf16x8*)p, *(const bf16x8*)(p + 16));
}

// B-fragment, 32x16 bf16, from an N-major (transposed) tile: lane holds
// column N=lane&15; lanes 0-15: K 0..7 then 16..23 ; lanes 16-31: K 8..15 then 24..31.
DEVFN bf16x16 load_frag_b(const __bf16* tileT, int ldn, int lane) {
  int col  = lane & 15;
  int koff = (lane & 16) ? 8 : 0;
  const __bf16* p = tileT + col * ldn + koff;
  return frag_cat(*(const bf16x8*)p, *(const bf16x8*)(p + 16));
}

// ---------------------------------------------------------------------------
// fp32 -> bf16 bulk conversion (4 elements / thread, n % 1024 == 0)
// ---------------------------------------------------------------------------
__global__ __launch_bounds__(256)
void cvt_f32_bf16_kernel(const float* __restrict__ in, __bf16* __restrict__ out) {
  size_t i = ((size_t)blockIdx.x * 256 + threadIdx.x) * 4;
  float4 f = *(const float4*)(in + i);
  bf16x4 h;
  h.x = (__bf16)f.x; h.y = (__bf16)f.y; h.z = (__bf16)f.z; h.w = (__bf16)f.w;
  *(bf16x4*)(out + i) = h;
}

// ---------------------------------------------------------------------------
// GEMM: Out[M,N] = A[M,K] * W[K,N], A and W bf16.
//  MODE 0: Out bf16 remapped to [B,H,S,Dh]  (projection kernels)
//  MODE 1: Out fp32 row-major               (output projection)
// Block: 256 threads = 8 waves; tile 128(M) x 128(N); k-step 32.
// Waves: 4 along M x 2 along N; each wave 32x64 = 2x4 WMMA tiles.
// ---------------------------------------------------------------------------
template<int MODE>
__global__ __launch_bounds__(256)
void wmma_gemm_kernel(const __bf16* __restrict__ A,
                      const __bf16* __restrict__ W,
                      void* __restrict__ Out,
                      int M, int N, int K)
{
  constexpr int LDA = 40;    // 128x32 tile, +8 pad (80B rows keep 16B align)
  constexpr int LDB = 40;    // 128(N) x 32(K) transposed tile
  __shared__ __bf16 lA[128 * LDA];
  __shared__ __bf16 lB[128 * LDB];

  const int tid  = threadIdx.x;
  const int lane = tid & 31;
  const int wv   = tid >> 5;
  const int bm   = blockIdx.x * 128;
  const int bn   = blockIdx.y * 128;
  const int wm   = (wv >> 1) * 32;   // 4 waves along M
  const int wn   = (wv & 1) * 64;    // 2 waves along N

  v8f acc[2][4] = {};

  for (int k0 = 0; k0 < K; k0 += 32) {
    // stage A tile 128x32 bf16 via async 16B copies (2 per thread)
    for (int i = tid; i < 512; i += 256) {
      int r = i >> 2, c8 = (i & 3) * 8;
      copy16_g2l(&lA[r * LDA + c8], A + (size_t)(bm + r) * K + k0 + c8);
    }
    // stage W tile 32x128, transposed to N-major lB[n][k] (16 elems/thread)
    for (int i = tid; i < 32 * 128; i += 256) {
      int kk = i >> 7, c = i & 127;                  // coalesced along c
      lB[c * LDB + kk] = W[(size_t)(k0 + kk) * N + bn + c];
    }
    if (k0 + 32 < K) {                               // prefetch next k-tile
      __builtin_prefetch(A + (size_t)(bm + (tid >> 1)) * K + k0 + 32, 0, 1);
      __builtin_prefetch(W + (size_t)(k0 + 32 + (tid >> 3)) * N + bn, 0, 1);
    }
    async_wait0();
    __syncthreads();

    bf16x16 af[2], bg[4];
    for (int mt = 0; mt < 2; ++mt)
      af[mt] = load_frag_a(lA + (wm + mt * 16) * LDA, LDA, lane);
    for (int nt = 0; nt < 4; ++nt)
      bg[nt] = load_frag_b(lB + (wn + nt * 16) * LDB, LDB, lane);

    for (int mt = 0; mt < 2; ++mt)
      for (int nt = 0; nt < 4; ++nt)
        acc[mt][nt] = __builtin_amdgcn_wmma_f32_16x16x32_bf16(
            false, af[mt], false, bg[nt], (short)0, acc[mt][nt], false, false);
    __syncthreads();
  }

  // Epilogue. C/D layout: lanes 0-15 -> M = vgpr, lanes 16-31 -> M = vgpr+8.
  const int halfsel = lane >> 4;
  const int nlane   = lane & 15;
  for (int mt = 0; mt < 2; ++mt)
    for (int nt = 0; nt < 4; ++nt)
      for (int i = 0; i < 8; ++i) {
        int r = bm + wm + mt * 16 + i + 8 * halfsel;
        int c = bn + wn + nt * 16 + nlane;
        float vvv = acc[mt][nt][i];
        if (MODE == 0) {
          // remap [row=b*S+s, col=h*64+d] -> bf16 [B,H,S,Dh]
          int b_ = r >> 11, s_ = r & (Sn - 1);
          int h_ = c >> 6,  d_ = c & (Dh - 1);
          ((__bf16*)Out)[(((size_t)b_ * Hn + h_) * Sn + s_) * Dh + d_] = (__bf16)vvv;
        } else {
          ((float*)Out)[(size_t)r * N + c] = vvv;
        }
      }
}

// ---------------------------------------------------------------------------
// Flash attention (causal). Grid: (B*H, S/128). Block: 256 thr = 8 waves.
// Wave w owns 16 query rows; loops over 64-key tiles up to the diagonal.
// Q/K/V are bf16 [B,H,S,64]; output bf16 [B,S,H*64].
// ---------------------------------------------------------------------------
__global__ __launch_bounds__(256)
void flash_attn_kernel(const __bf16* __restrict__ Q,
                       const __bf16* __restrict__ Kb,
                       const __bf16* __restrict__ Vb,
                       __bf16* __restrict__ O)
{
  constexpr int LDK = 72;                  // 64 + 8 pad (144B rows, 16B align)
  __shared__ __bf16 lK[64 * LDK];          // [key][d]   (B-frag for Q*K^T)
  __shared__ __bf16 lV[64 * LDK];          // [d][key]   (B-frag for P*V)
  __shared__ __bf16 lP[8][16 * LDK];       // wave-private C->A layout bounce

  const int tid  = threadIdx.x;
  const int lane = tid & 31;
  const int wv   = tid >> 5;
  const int bh   = blockIdx.x;             // b*H + h
  const int qb   = blockIdx.y;
  const int b_   = bh >> 4;
  const int h_   = bh & 15;
  const size_t head = (size_t)bh * Sn * Dh;
  const int qrow0 = qb * 128 + wv * 16;
  const int myrow = 8 * (lane >> 4);       // C-layout M offset for this lane

  // Q fragments resident in VGPRs: 16 rows x 64 d = two 16x32 A-frags
  bf16x16 aQ[2];
  {
    int row  = lane & 15;
    int koff = (lane & 16) ? 8 : 0;
    const __bf16* qp = Q + head + (size_t)(qrow0 + row) * Dh;
    for (int dt = 0; dt < 2; ++dt) {
      const __bf16* p = qp + dt * 32 + koff;
      aQ[dt] = frag_cat(*(const bf16x8*)p, *(const bf16x8*)(p + 16));
    }
  }

  v8f accO[4] = {};
  float mrow[8], lrow[8];
  for (int i = 0; i < 8; ++i) { mrow[i] = -1e30f; lrow[i] = 0.0f; }

  const float cexp = 0.04508422f;          // log2(e) / sqrt(QK)=32

  const int jend = (qb * 128 + 127) >> 6;  // causal: skip blocks above diag
  for (int j = 0; j <= jend; ++j) {
    const int kbase = j * 64;

    // stage K tile [64 keys][64 d] via async 16B copies (2 per thread)
    for (int i = tid; i < 512; i += 256) {
      int r = i >> 3, c8 = (i & 7) * 8;
      copy16_g2l(&lK[r * LDK + c8], Kb + head + (size_t)(kbase + r) * Dh + c8);
    }
    // stage V transposed [d][key]
    for (int i = tid; i < 4096; i += 256) {
      int r = i >> 6, c = i & 63;          // coalesced read along d
      lV[c * LDK + r] = Vb[head + (size_t)(kbase + r) * Dh + c];
    }
    async_wait0();
    __syncthreads();

    // scores S = Q K^T : 16 rows x 64 keys (4 N-tiles, 2 k-steps over d)
    v8f sc[4];
    for (int nt = 0; nt < 4; ++nt) {
      v8f z = {};
      bf16x16 bk0 = load_frag_b(lK + nt * 16 * LDK, LDK, lane);        // d 0..31
      bf16x16 bk1 = load_frag_b(lK + nt * 16 * LDK + 32, LDK, lane);   // d 32..63
      z = __builtin_amdgcn_wmma_f32_16x16x32_bf16(false, aQ[0], false, bk0,
                                                  (short)0, z, false, false);
      z = __builtin_amdgcn_wmma_f32_16x16x32_bf16(false, aQ[1], false, bk1,
                                                  (short)0, z, false, false);
      sc[nt] = z;
    }

    // scale into exp2 domain + causal mask (only needed near the diagonal)
    if (kbase + 63 > qrow0) {
      for (int nt = 0; nt < 4; ++nt) {
        int keyg = kbase + nt * 16 + (lane & 15);
        for (int i = 0; i < 8; ++i) {
          int qg = qrow0 + i + myrow;
          float t = sc[nt][i] * cexp;
          sc[nt][i] = (keyg > qg) ? -1e30f : t;
        }
      }
    } else {
      for (int nt = 0; nt < 4; ++nt)
        for (int i = 0; i < 8; ++i)
          sc[nt][i] = sc[nt][i] * cexp;
    }

    // online softmax: row stats live across 16 lanes of each half-wave
    float mloc[8];
    for (int i = 0; i < 8; ++i) {
      float mv = sc[0][i];
      mv = fmaxf(mv, sc[1][i]); mv = fmaxf(mv, sc[2][i]); mv = fmaxf(mv, sc[3][i]);
      mloc[i] = mv;
    }
    for (int off = 1; off < 16; off <<= 1)
      for (int i = 0; i < 8; ++i)
        mloc[i] = fmaxf(mloc[i], __shfl_xor(mloc[i], off, 32));

    float alpha[8], rsum[8];
    for (int i = 0; i < 8; ++i) {
      float mn = fmaxf(mrow[i], mloc[i]);
      alpha[i] = exp2f(mrow[i] - mn);
      mrow[i]  = mn;
      rsum[i]  = 0.0f;
    }
    for (int nt = 0; nt < 4; ++nt)
      for (int i = 0; i < 8; ++i) {
        float p = exp2f(sc[nt][i] - mrow[i]);
        sc[nt][i] = p;
        rsum[i] += p;
      }
    for (int off = 1; off < 16; off <<= 1)
      for (int i = 0; i < 8; ++i)
        rsum[i] += __shfl_xor(rsum[i], off, 32);
    for (int i = 0; i < 8; ++i) lrow[i] = lrow[i] * alpha[i] + rsum[i];
    for (int t = 0; t < 4; ++t)
      for (int i = 0; i < 8; ++i)
        accO[t][i] *= alpha[i];

    // bounce P through wave-private LDS: C-layout -> A-layout, fp32 -> bf16
    __bf16* pb = lP[wv];
    for (int nt = 0; nt < 4; ++nt)
      for (int i = 0; i < 8; ++i)
        pb[(i + myrow) * LDK + nt * 16 + (lane & 15)] = (__bf16)sc[nt][i];

    // O += P @ V  (k-dim = 64 keys: 2 k-steps; N = 64 d: 4 tiles)
    for (int kt = 0; kt < 2; ++kt) {
      bf16x16 aP = load_frag_a(pb + kt * 32, LDK, lane);
      for (int t = 0; t < 4; ++t) {
        bf16x16 bV = load_frag_b(lV + t * 16 * LDK + kt * 32, LDK, lane);
        accO[t] = __builtin_amdgcn_wmma_f32_16x16x32_bf16(
            false, aP, false, bV, (short)0, accO[t], false, false);
      }
    }
    __syncthreads();   // before next iteration restages lK/lV
  }

  // normalize + store bf16 at [b, s, h*64+d] (row-major for final GEMM)
  for (int i = 0; i < 8; ++i) lrow[i] = 1.0f / lrow[i];
  for (int t = 0; t < 4; ++t)
    for (int i = 0; i < 8; ++i) {
      int qg = qrow0 + i + myrow;
      int dg = t * 16 + (lane & 15);
      O[((size_t)b_ * Sn + qg) * (Hn * Dh) + h_ * Dh + dg] =
          (__bf16)(accO[t][i] * lrow[i]);
    }
}

// ---------------------------------------------------------------------------
extern "C" void kernel_launch(void* const* d_in, const int* in_sizes, int n_in,
                              void* d_out, int out_size, void* d_ws, size_t ws_size,
                              hipStream_t stream) {
  (void)in_sizes; (void)n_in; (void)out_size; (void)ws_size;
  const float* query = (const float*)d_in[0];
  const float* key_  = (const float*)d_in[1];
  const float* value = (const float*)d_in[2];
  // d_in[3] is the causal mask; implemented analytically in the kernel.
  const float* Wq = (const float*)d_in[4];
  const float* Wk = (const float*)d_in[5];
  const float* Wv = (const float*)d_in[6];
  const float* Wo = (const float*)d_in[7];

  const int M = Bn * Sn, N = En, K = En;                 // 8192, 1024, 1024
  const size_t headElems = (size_t)Bn * Hn * Sn * Dh;    // 8.39M

  __bf16* Xbf = (__bf16*)d_ws;            // staged bf16 activations [M,K]
  __bf16* Wbf = Xbf + (size_t)M * K;      // staged bf16 weights [K,N]
  __bf16* qws = Wbf + (size_t)K * N;      // q [B,H,S,64]
  __bf16* kws = qws + headElems;          // k
  __bf16* vws = kws + headElems;          // v
  __bf16* aws = vws + headElems;          // attn out [B,S,1024]

  dim3 blk(256);
  dim3 gCvtX((unsigned)((size_t)M * K / 1024));          // 8192 blocks
  dim3 gCvtW((unsigned)((size_t)K * N / 1024));          // 1024 blocks
  dim3 gGemm(M / 128, N / 128);                          // 64 x 8
  dim3 gAttn(Bn * Hn, Sn / 128);                         // 64 x 16

  // q = query*Wq
  cvt_f32_bf16_kernel<<<gCvtX, blk, 0, stream>>>(query, Xbf);
  cvt_f32_bf16_kernel<<<gCvtW, blk, 0, stream>>>(Wq, Wbf);
  wmma_gemm_kernel<0><<<gGemm, blk, 0, stream>>>(Xbf, Wbf, qws, M, N, K);
  // k = key*Wk
  cvt_f32_bf16_kernel<<<gCvtX, blk, 0, stream>>>(key_, Xbf);
  cvt_f32_bf16_kernel<<<gCvtW, blk, 0, stream>>>(Wk, Wbf);
  wmma_gemm_kernel<0><<<gGemm, blk, 0, stream>>>(Xbf, Wbf, kws, M, N, K);
  // v = value*Wv
  cvt_f32_bf16_kernel<<<gCvtX, blk, 0, stream>>>(value, Xbf);
  cvt_f32_bf16_kernel<<<gCvtW, blk, 0, stream>>>(Wv, Wbf);
  wmma_gemm_kernel<0><<<gGemm, blk, 0, stream>>>(Xbf, Wbf, vws, M, N, K);
  // attention
  flash_attn_kernel<<<gAttn, blk, 0, stream>>>(qws, kws, vws, aws);
  // out = attn*Wo (fp32)
  cvt_f32_bf16_kernel<<<gCvtW, blk, 0, stream>>>(Wo, Wbf);
  wmma_gemm_kernel<1><<<gGemm, blk, 0, stream>>>(aws, Wbf, d_out, M, N, K);
}
